// AtomicDipolesMACE_43293270343921
// MI455X (gfx1250) — matrix operand order repaired and checked by reading, not compiled, not run on hardware
//
#include <hip/hip_runtime.h>
#include <hip/hip_bf16.h>
#include <math.h>

typedef __attribute__((ext_vector_type(16))) _Float16 v16h;
typedef __attribute__((ext_vector_type(8)))  float    v8f;

#define NN   8192
#define EE   131072
#define CC   64
#define NELN 10
#define NBN  8
#define GG   32
#define HH   16
#define PI_F 3.14159265358979f
#define RMAXF 5.0f

__constant__ int c_lof[16]   = {0,1,1,1,2,2,2,2,2,3,3,3,3,3,3,3};
__constant__ int c_kstart[4] = {0,1,4,9};
__constant__ int c_kcount[4] = {1,3,5,7};

__device__ __forceinline__ float siluf(float x) { return x / (1.f + __expf(-x)); }

// ---------------- pack a K x ncols f32 matrix into f16 WMMA-B lane layout ----------------
// dst element t = ((nt*2 + chunk)*32 + lane), 16 halves each:
//   dst[t][i] = W[(chunk*32 + (lane>=16?8:0) + i + (i>=8?8:0)) * ncols + nt*16 + (lane&15)]
__global__ void mace_pack_b(const float* __restrict__ W, _Float16* __restrict__ P, int ncols) {
  int t = blockIdx.x*blockDim.x + threadIdx.x;
  int lane  = t & 31;
  int chunk = (t >> 5) & 1;
  int nt    = t >> 6;
  if (nt >= ncols/16) return;
  int n  = nt*16 + (lane & 15);
  int kb = chunk*32 + ((lane >= 16) ? 8 : 0);
  _Float16* dst = P + (size_t)t*16;
  #pragma unroll
  for (int i = 0; i < 16; ++i) dst[i] = (_Float16)W[(kb + i + ((i >= 8) ? 8 : 0))*ncols + n];
}

// ---------------- node embedding + element index ----------------
__global__ __launch_bounds__(64) void mace_embed(const float* __restrict__ attrs,
                                                 const float* __restrict__ Wemb,
                                                 float* __restrict__ h0, int* __restrict__ elem) {
  int n = blockIdx.x, d = threadIdx.x;
  __shared__ int s_el;
  if (d == 0) {
    int el = 0;
    for (int k = 0; k < NELN; ++k) if (attrs[n*NELN + k] > 0.5f) el = k;
    s_el = el; elem[n] = el;
  }
  __syncthreads();
  h0[(size_t)n*CC + d] = Wemb[s_el*CC + d];
}

// ---------------- edge geometry: Y (16 sph harm) + radial basis ----------------
__global__ void mace_edge_geom(const float* __restrict__ pos, const float* __restrict__ shifts,
                               const int* __restrict__ snd, const int* __restrict__ rcv,
                               float* __restrict__ Yb, float* __restrict__ rbb) {
  int e = blockIdx.x*blockDim.x + threadIdx.x;
  if (e >= EE) return;
  int s = snd[e], t = rcv[e];
  float vx = pos[t*3+0]-pos[s*3+0]+shifts[e*3+0];
  float vy = pos[t*3+1]-pos[s*3+1]+shifts[e*3+1];
  float vz = pos[t*3+2]-pos[s*3+2]+shifts[e*3+2];
  float r = sqrtf(vx*vx+vy*vy+vz*vz+1e-12f);
  float ir = 1.f/r;
  float x = vx*ir, y = vy*ir, z = vz*ir;
  const float s3=1.73205081f, s15=3.87298335f;
  float x2=x*x, y2=y*y, z2=z*z;
  float Y[16];
  Y[0]=1.f; Y[1]=s3*x; Y[2]=s3*y; Y[3]=s3*z;
  Y[4]=s15*x*y; Y[5]=s15*y*z; Y[6]=1.11803399f*(3.f*z2-1.f);
  Y[7]=s15*x*z; Y[8]=1.93649167f*(x2-y2);
  Y[9]=2.09165007f*y*(3.f*x2-y2);
  Y[10]=10.24695077f*x*y*z;
  Y[11]=1.62018517f*y*(5.f*z2-1.f);
  Y[12]=1.32287566f*z*(5.f*z2-3.f);
  Y[13]=1.62018517f*x*(5.f*z2-1.f);
  Y[14]=5.12347538f*z*(x2-y2);
  Y[15]=2.09165007f*x*(x2-3.f*y2);
  #pragma unroll
  for (int k = 0; k < 16; ++k) Yb[(size_t)e*16 + k] = Y[k];
  float xr = r*(1.f/RMAXF);
  float x2r = xr*xr, x4r = x2r*x2r, x5 = x4r*xr, x6 = x5*xr, x7 = x6*xr;
  float env = 1.f - 21.f*x5 + 35.f*x6 - 15.f*x7;
  if (xr >= 1.f) env = 0.f;
  const float c0 = 0.63245553f; // sqrt(2/5)
  #pragma unroll
  for (int b = 0; b < NBN; ++b)
    rbb[(size_t)e*NBN + b] = c0 * sinf((b+1)*PI_F*r*(1.f/RMAXF)) * ir * env;
}

// ---------------- edge radial MLP (WMMA) + message + atomic scatter ----------------
// one wave per 16-edge tile; 4 tiles per 128-thread block
__global__ __launch_bounds__(128) void mace_radial_scatter(
    const float* __restrict__ rbb, const float* __restrict__ Yb,
    const int* __restrict__ snd, const int* __restrict__ rcv,
    const float* __restrict__ feat,                     // h0 (layer1) / out0 (layer2), N x 64
    const float* __restrict__ Wa,                       // 8 x 64
    const _Float16* __restrict__ WbP,                   // packed 64 x 256 B-layout
    float* __restrict__ Aout)                           // N x 16 x 64
{
  __shared__ _Float16 s_hid[4][16*CC];
  int w = threadIdx.x >> 5, lane = threadIdx.x & 31;
  int e0 = (blockIdx.x*4 + w) * 16;

  // speculative prefetch of this tile's Y block (used in scatter phase)
  __builtin_prefetch(&Yb[(size_t)e0*16], 0, 0);

  // phase 1: hidden = silu(rb @ Wa), 16x64, scalar (tiny K=8)
  for (int idx = lane; idx < 16*CC; idx += 32) {
    int er = idx >> 6, c = idx & 63;
    const float* rbp = &rbb[(size_t)(e0+er)*NBN];
    float acc = 0.f;
    #pragma unroll
    for (int b = 0; b < NBN; ++b) acc += rbp[b] * Wa[b*CC + c];
    s_hid[w][er*CC + c] = (_Float16)siluf(acc);
  }
  __syncthreads();

  // phase 2: wr = hidden(16x64) @ Wb(64x256) via v_wmma_f32_16x16x32_f16
  // A operand: contiguous 16B runs in LDS -> ds_load_b128
  int m    = lane & 15;
  int kofs = (lane >= 16) ? 8 : 0;
  const _Float16* hb = &s_hid[w][m*CC + kofs];
  union { float4 q[2]; v16h v; } ua0, ua1;
  ua0.q[0] = *(const float4*)(hb);          // halves k .. k+7   (chunk 0)
  ua0.q[1] = *(const float4*)(hb + 16);     // halves k+16..k+23
  ua1.q[0] = *(const float4*)(hb + 32);     // chunk 1
  ua1.q[1] = *(const float4*)(hb + 48);
  v16h a0 = ua0.v, a1 = ua1.v;

  int colq = lane & 15;
  for (int nt = 0; nt < 16; ++nt) {
    // B operand: pre-packed per-lane v16h -> two global_load_b128 each
    v16h b0 = *(const v16h*)(WbP + (size_t)((nt*2 + 0)*32 + lane)*16);
    v16h b1 = *(const v16h*)(WbP + (size_t)((nt*2 + 1)*32 + lane)*16);
    v8f acc = {};
    acc = __builtin_amdgcn_wmma_f32_16x16x32_f16(false, a0, false, b0, (short)0, acc, false, false);
    acc = __builtin_amdgcn_wmma_f32_16x16x32_f16(false, a1, false, b1, (short)0, acc, false, false);
    // scatter: M[e,k,c] = feat[snd[e],c]*Y[e,k]*wr[e, lof(k)*64+c]; A is L2-resident
    int n = nt*16 + colq;
    int l = n >> 6, c = n & 63;
    int ks = c_kstart[l], kc = c_kcount[l];
    #pragma unroll
    for (int r = 0; r < 8; ++r) {
      int e = e0 + r + ((lane >= 16) ? 8 : 0);
      int sn = snd[e], rv = rcv[e];
      float wrv = acc[r] * feat[(size_t)sn*CC + c];
      for (int kk = 0; kk < kc; ++kk) {
        int k = ks + kk;
        atomicAdd(&Aout[(size_t)rv*1024 + k*CC + c], Yb[(size_t)e*16 + k] * wrv);
      }
    }
  }
}

// ---------------- h[n,k,:] = (A[n,k,:]/AVG) @ Wmix[lof(k)]  (WMMA, in place) ----------------
__global__ __launch_bounds__(128) void mace_mix(const float* __restrict__ A,
                                                const _Float16* __restrict__ WmixP, // packed 4 x (64x64)
                                                float* __restrict__ H) {
  int w = threadIdx.x >> 5, lane = threadIdx.x & 31;
  int wg = blockIdx.x*4 + w;
  int ntile = wg >> 4, k = wg & 15;
  int l = c_lof[k];
  int m = lane & 15;
  int kofs = (lane >= 16) ? 8 : 0;
  const float* arow = &A[(size_t)(ntile*16 + m)*1024 + k*CC + kofs];
  const float sc = 1.0f/16.0f; // 1/AVG

  // A operand: 8 x global_load_b128 + packed converts
  float4 q0 = *(const float4*)(arow);        // chunk0 halves 0-3
  float4 q1 = *(const float4*)(arow + 4);    // chunk0 halves 4-7
  float4 q2 = *(const float4*)(arow + 16);   // chunk0 halves 8-11
  float4 q3 = *(const float4*)(arow + 20);   // chunk0 halves 12-15
  float4 r0 = *(const float4*)(arow + 32);   // chunk1
  float4 r1 = *(const float4*)(arow + 36);
  float4 r2 = *(const float4*)(arow + 48);
  float4 r3 = *(const float4*)(arow + 52);
  float av0[16] = {q0.x,q0.y,q0.z,q0.w, q1.x,q1.y,q1.z,q1.w,
                   q2.x,q2.y,q2.z,q2.w, q3.x,q3.y,q3.z,q3.w};
  float av1[16] = {r0.x,r0.y,r0.z,r0.w, r1.x,r1.y,r1.z,r1.w,
                   r2.x,r2.y,r2.z,r2.w, r3.x,r3.y,r3.z,r3.w};
  v16h a0, a1;
  #pragma unroll
  for (int i = 0; i < 16; ++i) {
    a0[i] = (_Float16)(av0[i]*sc);
    a1[i] = (_Float16)(av1[i]*sc);
  }

  int colq = lane & 15;
  const _Float16* Bp = WmixP + (size_t)l*4096;   // 4 tiles * 2 chunks * 32 lanes * 16 halves
  #pragma unroll
  for (int dt = 0; dt < 4; ++dt) {
    v16h b0 = *(const v16h*)(Bp + (size_t)((dt*2 + 0)*32 + lane)*16);
    v16h b1 = *(const v16h*)(Bp + (size_t)((dt*2 + 1)*32 + lane)*16);
    v8f acc = {};
    acc = __builtin_amdgcn_wmma_f32_16x16x32_f16(false, a0, false, b0, (short)0, acc, false, false);
    acc = __builtin_amdgcn_wmma_f32_16x16x32_f16(false, a1, false, b1, (short)0, acc, false, false);
    int n = dt*16 + colq;
    #pragma unroll
    for (int r = 0; r < 8; ++r) {
      int row = ntile*16 + r + ((lane >= 16) ? 8 : 0);
      H[(size_t)row*1024 + k*CC + n] = acc[r];
    }
  }
}

// ---------------- layer-1 node update ----------------
__global__ __launch_bounds__(64) void mace_node1(
    const float* __restrict__ h0, const float* __restrict__ H,
    const float* __restrict__ Wsc1, const float* __restrict__ Wps, const float* __restrict__ Wpv,
    const float* __restrict__ Wp0, const float* __restrict__ Wp1, const float* __restrict__ wread,
    const int* __restrict__ elem, float* __restrict__ out0, float* __restrict__ out1,
    float* __restrict__ dip) {
  int n = blockIdx.x, d = threadIdx.x;
  int el = elem[n];
  __shared__ float s_h0[CC], s_m0[CC], s_t[3*CC], s_o1[3*CC];
  s_h0[d] = h0[(size_t)n*CC + d];
  float sv = H[(size_t)n*1024 + d];
  float ws0 = Wps[(0*NELN+el)*CC+d], ws1 = Wps[(1*NELN+el)*CC+d], ws2 = Wps[(2*NELN+el)*CC+d];
  float wv0 = Wpv[(0*NELN+el)*CC+d], wv1 = Wpv[(1*NELN+el)*CC+d], wv2 = Wpv[(2*NELN+el)*CC+d];
  s_m0[d] = sv*(ws0 + sv*(ws1 + sv*ws2));
  float gv = wv0 + sv*(wv1 + sv*wv2);
  #pragma unroll
  for (int mm = 0; mm < 3; ++mm) s_t[mm*CC+d] = H[(size_t)n*1024 + (1+mm)*CC + d] * gv;
  __syncthreads();
  float acc = 0.f, sc = 0.f;
  for (int c = 0; c < CC; ++c) {
    acc += s_m0[c]*Wp0[c*CC+d];
    sc  += s_h0[c]*Wsc1[(size_t)el*4096 + c*CC + d];
  }
  out0[(size_t)n*CC + d] = acc + sc;
  #pragma unroll
  for (int mm = 0; mm < 3; ++mm) {
    float a = 0.f;
    for (int c = 0; c < CC; ++c) a += s_t[mm*CC+c]*Wp1[c*CC+d];
    s_o1[mm*CC+d] = a;
    out1[(size_t)n*192 + mm*CC + d] = a;
  }
  __syncthreads();
  if (d < 3) {
    float dd = 0.f;
    for (int j = 0; j < CC; ++j) dd += s_o1[d*CC+j]*wread[j];
    dip[(size_t)n*3 + d] = dd;
  }
}

// ---------------- layer-2 node update + gated readout ----------------
__global__ __launch_bounds__(64) void mace_node2(
    const float* __restrict__ out1, const float* __restrict__ H,
    const float* __restrict__ Wsc2, const float* __restrict__ Wpr2, const float* __restrict__ Wp2,
    const float* __restrict__ Wv, const float* __restrict__ Wg1, const float* __restrict__ bg1,
    const float* __restrict__ Wg2, const float* __restrict__ bg2, const float* __restrict__ wread2,
    const int* __restrict__ elem, float* __restrict__ dip) {
  int n = blockIdx.x, d = threadIdx.x;
  int el = elem[n];
  __shared__ float s_t[3*CC], s_o1[3*CC], s_o2[3*CC], s_vh[48], s_inv[HH], s_u[HH], s_a[HH];
  float s2 = H[(size_t)n*1024 + d];
  float w0 = Wpr2[(0*NELN+el)*CC+d], w1 = Wpr2[(1*NELN+el)*CC+d], w2 = Wpr2[(2*NELN+el)*CC+d];
  float gv2 = w0 + s2*(w1 + s2*w2);
  #pragma unroll
  for (int mm = 0; mm < 3; ++mm) {
    s_t[mm*CC+d]  = H[(size_t)n*1024 + (1+mm)*CC + d] * gv2;
    s_o1[mm*CC+d] = out1[(size_t)n*192 + mm*CC + d];
  }
  __syncthreads();
  #pragma unroll
  for (int mm = 0; mm < 3; ++mm) {
    float a = 0.f, b = 0.f;
    for (int c = 0; c < CC; ++c) {
      a += s_t[mm*CC+c]*Wp2[c*CC+d];
      b += s_o1[mm*CC+c]*Wsc2[(size_t)el*4096 + c*CC + d];
    }
    s_o2[mm*CC+d] = a + b;
  }
  __syncthreads();
  if (d < 48) {
    int mm = d >> 4, hh = d & 15;
    float a = 0.f;
    for (int c = 0; c < CC; ++c) a += s_o2[mm*CC+c]*Wv[c*HH+hh];
    s_vh[d] = a;
  }
  __syncthreads();
  if (d < HH) {
    float v0 = s_vh[d], v1 = s_vh[16+d], v2 = s_vh[32+d];
    s_inv[d] = sqrtf(v0*v0 + v1*v1 + v2*v2 + 1e-12f);
  }
  __syncthreads();
  if (d < HH) {
    float a = bg1[d];
    for (int j = 0; j < HH; ++j) a += s_inv[j]*Wg1[j*HH+d];
    s_u[d] = siluf(a);
  }
  __syncthreads();
  if (d < HH) {
    float a = bg2[d];
    for (int j = 0; j < HH; ++j) a += s_u[j]*Wg2[j*HH+d];
    s_a[d] = a;
  }
  __syncthreads();
  if (d < 3) {
    float dd = 0.f;
    for (int h = 0; h < HH; ++h) dd += s_vh[d*HH+h]*s_a[h]*wread2[h];
    dip[(size_t)n*3 + d] += dd;
  }
}

// ---------------- per-graph totals ----------------
__global__ void mace_total(const float* __restrict__ dip, const float* __restrict__ charges,
                           const float* __restrict__ pos, const int* __restrict__ batch,
                           float* __restrict__ total) {
  int n = blockIdx.x*blockDim.x + threadIdx.x;
  if (n >= NN) return;
  int g = batch[n];
  float q = charges[n];
  #pragma unroll
  for (int mm = 0; mm < 3; ++mm)
    atomicAdd(&total[g*3+mm], dip[(size_t)n*3+mm] + q*pos[(size_t)n*3+mm]);
}

extern "C" void kernel_launch(void* const* d_in, const int* in_sizes, int n_in,
                              void* d_out, int out_size, void* d_ws, size_t ws_size,
                              hipStream_t stream) {
  const float* node_attrs = (const float*)d_in[0];
  const float* positions  = (const float*)d_in[1];
  const float* shifts     = (const float*)d_in[2];
  const float* charges    = (const float*)d_in[3];
  const float* W_embed    = (const float*)d_in[4];
  const float* Wr1a       = (const float*)d_in[5];
  const float* Wr1b       = (const float*)d_in[6];
  const float* Wmix1      = (const float*)d_in[7];
  const float* Wsc1       = (const float*)d_in[8];
  const float* Wprod1s    = (const float*)d_in[9];
  const float* Wprod1v    = (const float*)d_in[10];
  const float* Wp1_0      = (const float*)d_in[11];
  const float* Wp1_1      = (const float*)d_in[12];
  const float* w_read1    = (const float*)d_in[13];
  const float* Wr2a       = (const float*)d_in[14];
  const float* Wr2b       = (const float*)d_in[15];
  const float* Wmix2      = (const float*)d_in[16];
  const float* Wsc2       = (const float*)d_in[17];
  const float* Wprod2     = (const float*)d_in[18];
  const float* Wp2        = (const float*)d_in[19];
  const float* Wv         = (const float*)d_in[20];
  const float* Wg1        = (const float*)d_in[21];
  const float* bg1        = (const float*)d_in[22];
  const float* Wg2        = (const float*)d_in[23];
  const float* bg2        = (const float*)d_in[24];
  const float* w_read2    = (const float*)d_in[25];
  const int*   edge_index = (const int*)d_in[26];
  const int*   batch      = (const int*)d_in[27];
  const int* snd = edge_index;
  const int* rcv = edge_index + EE;

  float* out   = (float*)d_out;
  float* total = out;            // G*3
  float* dip   = out + GG*3;     // N*3

  // workspace carve
  float* Abuf = (float*)d_ws;                            // N*16*C  (A, then h in place)
  float* h0   = Abuf + (size_t)NN*16*CC;                 // N*C
  float* Yb   = h0   + (size_t)NN*CC;                    // E*16
  float* rbb  = Yb   + (size_t)EE*16;                    // E*8
  float* out0 = rbb  + (size_t)EE*NBN;                   // N*C
  float* out1 = out0 + (size_t)NN*CC;                    // N*3*C
  int*   elem = (int*)(out1 + (size_t)NN*3*CC);          // N
  _Float16* WbP1   = (_Float16*)(elem + NN);             // 64*256 halves
  _Float16* WbP2   = WbP1  + 64*256;                     // 64*256 halves
  _Float16* WmixP1 = WbP2  + 64*256;                     // 4*64*64 halves
  _Float16* WmixP2 = WmixP1 + 4*64*64;                   // 4*64*64 halves

  hipMemsetAsync(total, 0, GG*3*sizeof(float), stream);

  // pre-pack f16 B operands (tiny, L2-resident, once per launch)
  mace_pack_b<<<4, 256, 0, stream>>>(Wr1b, WbP1, 256);
  mace_pack_b<<<4, 256, 0, stream>>>(Wr2b, WbP2, 256);
  for (int l = 0; l < 4; ++l) {
    mace_pack_b<<<1, 256, 0, stream>>>(Wmix1 + (size_t)l*4096, WmixP1 + (size_t)l*4096, 64);
    mace_pack_b<<<1, 256, 0, stream>>>(Wmix2 + (size_t)l*4096, WmixP2 + (size_t)l*4096, 64);
  }

  mace_embed<<<NN, 64, 0, stream>>>(node_attrs, W_embed, h0, elem);
  mace_edge_geom<<<EE/256, 256, 0, stream>>>(positions, shifts, snd, rcv, Yb, rbb);

  // layer 1
  hipMemsetAsync(Abuf, 0, (size_t)NN*16*CC*sizeof(float), stream);
  mace_radial_scatter<<<EE/64, 128, 0, stream>>>(rbb, Yb, snd, rcv, h0, Wr1a, WbP1, Abuf);
  mace_mix<<<(NN/16*16)/4, 128, 0, stream>>>(Abuf, WmixP1, Abuf);
  mace_node1<<<NN, 64, 0, stream>>>(h0, Abuf, Wsc1, Wprod1s, Wprod1v, Wp1_0, Wp1_1, w_read1,
                                    elem, out0, out1, dip);

  // layer 2
  hipMemsetAsync(Abuf, 0, (size_t)NN*16*CC*sizeof(float), stream);
  mace_radial_scatter<<<EE/64, 128, 0, stream>>>(rbb, Yb, snd, rcv, out0, Wr2a, WbP2, Abuf);
  mace_mix<<<(NN/16*16)/4, 128, 0, stream>>>(Abuf, WmixP2, Abuf);
  mace_node2<<<NN, 64, 0, stream>>>(out1, Abuf, Wsc2, Wprod2, Wp2, Wv, Wg1, bg1, Wg2, bg2,
                                    w_read2, elem, dip);

  mace_total<<<NN/256, 256, 0, stream>>>(dip, charges, positions, batch, total);
}
